// IntraPatchGraphAggregator_43301860278555
// MI455X (gfx1250) — matrix operand order repaired and checked by reading, not compiled, not run on hardware
//
#include <hip/hip_runtime.h>
#include <math.h>

// ---------------- problem constants (from reference) ----------------
#define N_PATCH 8192
#define LSEQ    64
#define DMODEL  256
#define NHEAD   4
#define DHEAD   64
#define TAU_F   300.0f

typedef _Float16 v16h __attribute__((ext_vector_type(16)));
typedef _Float16 v8h  __attribute__((ext_vector_type(8)));
typedef float    v8f  __attribute__((ext_vector_type(8)));

// D = A(16x32 f16) * B(32x16 f16) + C(16x16 f32)
static __device__ __forceinline__ v8f wmma_f16(v16h a, v16h b, v8f c) {
  return __builtin_amdgcn_wmma_f32_16x16x32_f16(false, a, false, b,
                                                (short)0, c, false, false);
}

// A fragment: 16x32 f16 tile from row-major storage (row stride in halves).
// CDNA5 layout: lane m=lane&15 holds row r0+m; half=lane>>4 selects
// K = {half*8 .. half*8+7, 16+half*8 .. 16+half*8+7}.
static __device__ __forceinline__ v16h load_a(const _Float16* base, int r0,
                                              int k0, int rowstride, int lane) {
  const int m = r0 + (lane & 15);
  const int hf = lane >> 4;
  const _Float16* row = base + m * rowstride + k0;
  v8h lo = *(const v8h*)(row + hf * 8);
  v8h hi = *(const v8h*)(row + 16 + hf * 8);
  v16h a;
#pragma unroll
  for (int i = 0; i < 8; ++i) { a[i] = lo[i]; a[i + 8] = hi[i]; }
  return a;
}

// B fragment: 32x16 f16 where B[k][n] = src[(n0+n)*rowstride + k0 + k].
// CDNA5 layout: lane holds column n0+(lane&15), half=lane>>4 selects
// K = {half*8 .. half*8+7, 16+half*8 .. 16+half*8+7}.
static __device__ __forceinline__ v16h load_bt(const _Float16* base, int n0,
                                               int k0, int rowstride, int lane) {
  const int nl = lane & 15;
  const int hf = lane >> 4;
  const _Float16* col = base + (n0 + nl) * rowstride + k0;
  v8h lo = *(const v8h*)(col + hf * 8);
  v8h hi = *(const v8h*)(col + 16 + hf * 8);
  v16h b;
#pragma unroll
  for (int i = 0; i < 8; ++i) { b[i] = lo[i]; b[i + 8] = hi[i]; }
  return b;
}

// ---------------- weight pre-conversion: fp32 -> f16 into workspace --------
__global__ void convert_weights_kernel(const float* __restrict__ Wq,
                                       const float* __restrict__ Wk,
                                       const float* __restrict__ Wv,
                                       const float* __restrict__ Wo,
                                       _Float16* __restrict__ w16) {
  const int i = blockIdx.x * blockDim.x + threadIdx.x;   // 0 .. 4*65536-1
  const int which = i >> 16;
  const int off   = i & 65535;
  const float* src = (which == 0) ? Wq : (which == 1) ? Wk
                   : (which == 2) ? Wv : Wo;
  w16[i] = (_Float16)src[off];
}

// ---------------- fused per-patch attention + pooling ----------------------
// one workgroup (8 wave32s) per patch; everything stays in LDS.
__global__ __launch_bounds__(256, 1)
void patch_attn_pool_kernel(const float* __restrict__ tokens,
                            const float* __restrict__ ts,
                            const float* __restrict__ mask,
                            const float* __restrict__ bq,
                            const float* __restrict__ bk,
                            const float* __restrict__ bv,
                            const float* __restrict__ bo,
                            const float* __restrict__ readout,
                            const _Float16* __restrict__ w16,
                            float* __restrict__ out) {
  extern __shared__ char smem[];
  _Float16* sT   = (_Float16*)(smem);            // 64x256 tokens  (reused: attn-out)
  _Float16* sQ   = (_Float16*)(smem + 32768);    // 64x256 Q*scale (reused: final out)
  _Float16* sK   = (_Float16*)(smem + 65536);    // 64x256 K row-major
  _Float16* sVt  = (_Float16*)(smem + 98304);    // 256x64 V transposed
  float*    sS   = (float*)   (smem + 131072);   // 64x64 raw scores (per head)
  _Float16* sP   = (_Float16*)(smem + 147456);   // 64x64 probabilities
  float*    st   = (float*)   (smem + 155648);   // 64 timestamps
  float*    smk  = (float*)   (smem + 155904);   // 64 mask
  float*    sZ   = (float*)   (smem + 156160);   // 4x64 pooling logits
  float*    sRed = (float*)   (smem + 157184);   // 4 x (max, sum)

  const int n    = blockIdx.x;
  const int tid  = threadIdx.x;
  const int lane = tid & 31;
  const int wave = tid >> 5;
  const int nl   = lane & 15;
  const int hf   = lane >> 4;
  const float scale = 0.125f;                    // 1/sqrt(DHEAD)

  const _Float16* wq16 = w16;
  const _Float16* wk16 = w16 + 65536;
  const _Float16* wv16 = w16 + 131072;
  const _Float16* wo16 = w16 + 196608;

  // ---- stage 0: tokens -> LDS (f32 -> f16), timestamps + mask ----
  const float* tokN = tokens + (size_t)n * (LSEQ * DMODEL);
  for (int i = tid; i < (LSEQ * DMODEL) / 4; i += 256) {
    float4 v = ((const float4*)tokN)[i];
    const int b = i * 4;
    sT[b + 0] = (_Float16)v.x; sT[b + 1] = (_Float16)v.y;
    sT[b + 2] = (_Float16)v.z; sT[b + 3] = (_Float16)v.w;
  }
  if (tid < LSEQ) {
    st[tid]  = ts[n * LSEQ + tid];
    smk[tid] = mask[n * LSEQ + tid];
  }
  __syncthreads();

  // ---- stage 1: Q/K/V projections  X(64x256) @ W^T(256x256) + b ----
  // 3 projections x 64 tiles of 16x16; each tile = 8 WMMAs over K=256.
  for (int job = wave; job < 192; job += 8) {
    const int p  = job >> 6;                 // 0=Q 1=K 2=V
    const int t  = job & 63;
    const int r0 = (t >> 4) * 16;
    const int c0 = (t & 15) * 16;
    const _Float16* W = (p == 0) ? wq16 : (p == 1) ? wk16 : wv16;
    __builtin_prefetch(W + c0 * DMODEL, 0, 3);      // global_prefetch_b8
    v8f acc = {};
#pragma unroll
    for (int kk = 0; kk < 8; ++kk) {
      v16h a = load_a(sT, r0, kk * 32, DMODEL, lane);
      v16h b = load_bt(W, c0, kk * 32, DMODEL, lane);
      acc = wmma_f16(a, b, acc);
    }
    const int j = c0 + nl;
    const float bias = (p == 0) ? bq[j] : (p == 1) ? bk[j] : bv[j];
    if (p == 0) {
#pragma unroll
      for (int r = 0; r < 8; ++r)
        sQ[(r0 + hf * 8 + r) * DMODEL + j] = (_Float16)((acc[r] + bias) * scale);
    } else if (p == 1) {
#pragma unroll
      for (int r = 0; r < 8; ++r)
        sK[(r0 + hf * 8 + r) * DMODEL + j] = (_Float16)(acc[r] + bias);
    } else {
#pragma unroll
      for (int r = 0; r < 8; ++r)
        sVt[j * LSEQ + (r0 + hf * 8 + r)] = (_Float16)(acc[r] + bias);
    }
  }
  __syncthreads();

  // ---- stage 2: per-head attention ----
  _Float16* sAO = sT;                         // tokens dead -> reuse as attn-out
  for (int h = 0; h < NHEAD; ++h) {
    const int koff = h * DHEAD;

    // S = (Q*scale) @ K^T : 16 tiles, K-depth 64 (2 WMMAs each)
    for (int t = wave; t < 16; t += 8) {
      const int r0 = (t >> 2) * 16, c0 = (t & 3) * 16;
      v8f acc = {};
#pragma unroll
      for (int kk = 0; kk < 2; ++kk) {
        v16h a = load_a(sQ, r0, koff + kk * 32, DMODEL, lane);
        v16h b = load_bt(sK, c0, koff + kk * 32, DMODEL, lane);
        acc = wmma_f16(a, b, acc);
      }
#pragma unroll
      for (int r = 0; r < 8; ++r)
        sS[(r0 + hf * 8 + r) * LSEQ + c0 + nl] = acc[r];
    }
    __syncthreads();

    // time-decay bias + pad mask + row softmax (4 lanes per row, shfl_xor)
    {
      const int row = tid >> 2, seg = tid & 3;
      const float ti = st[row];
      float vals[16];
      float mx = -3.0e38f;
#pragma unroll
      for (int c = 0; c < 16; ++c) {
        const int col = seg * 16 + c;
        float s = sS[row * LSEQ + col];
        const float dt = fabsf(ti - st[col]);
        s += logf(__expf(-dt * (1.0f / TAU_F)) + 1e-12f);
        if (smk[col] <= 0.0f) s = -1e9f;
        vals[c] = s;
        mx = fmaxf(mx, s);
      }
      mx = fmaxf(mx, __shfl_xor(mx, 1, 32));
      mx = fmaxf(mx, __shfl_xor(mx, 2, 32));
      float sum = 0.0f;
#pragma unroll
      for (int c = 0; c < 16; ++c) { vals[c] = __expf(vals[c] - mx); sum += vals[c]; }
      sum += __shfl_xor(sum, 1, 32);
      sum += __shfl_xor(sum, 2, 32);
      const float inv = 1.0f / sum;
#pragma unroll
      for (int c = 0; c < 16; ++c)
        sP[row * LSEQ + seg * 16 + c] = (_Float16)(vals[c] * inv);
    }
    __syncthreads();

    // O_h = P @ V_h  (V stored transposed -> contiguous B fragments)
    for (int t = wave; t < 16; t += 8) {
      const int r0 = (t >> 2) * 16, c0 = (t & 3) * 16;
      v8f acc = {};
#pragma unroll
      for (int kk = 0; kk < 2; ++kk) {
        v16h a = load_a(sP, r0, kk * 32, LSEQ, lane);
        v16h b = load_bt(sVt, koff + c0, kk * 32, LSEQ, lane);
        acc = wmma_f16(a, b, acc);
      }
#pragma unroll
      for (int r = 0; r < 8; ++r)
        sAO[(r0 + hf * 8 + r) * DMODEL + koff + c0 + nl] = (_Float16)acc[r];
    }
    __syncthreads();
  }

  // ---- stage 3: output projection  F = (AO @ Wo^T + bo) * mask ----
  _Float16* sF = sQ;                          // Q dead -> reuse as final out
  for (int t = wave; t < 64; t += 8) {
    const int r0 = (t >> 4) * 16, c0 = (t & 15) * 16;
    __builtin_prefetch(wo16 + c0 * DMODEL, 0, 3);
    v8f acc = {};
#pragma unroll
    for (int kk = 0; kk < 8; ++kk) {
      v16h a = load_a(sAO, r0, kk * 32, DMODEL, lane);
      v16h b = load_bt(wo16, c0, kk * 32, DMODEL, lane);
      acc = wmma_f16(a, b, acc);
    }
    const int j = c0 + nl;
    const float bj = bo[j];
#pragma unroll
    for (int r = 0; r < 8; ++r) {
      const int row = r0 + hf * 8 + r;
      sF[row * DMODEL + j] = (_Float16)((acc[r] + bj) * smk[row]);
    }
  }
  __syncthreads();

  // ---- stage 4: learned-query pooling over timesteps ----
  {
    const int h = tid >> 6, l = tid & 63;     // one (head, key-row) per thread
    const float* rh = readout + h * DHEAD;
    float z = 0.0f;
    for (int d0 = 0; d0 < DHEAD; ++d0)
      z += rh[d0] * (float)sF[l * DMODEL + h * DHEAD + d0];
    z *= scale;
    if (smk[l] <= 0.0f) z = -1e9f;
    sZ[h * LSEQ + l] = z;
  }
  __syncthreads();
  if (tid < NHEAD) {
    float mx = -3.0e38f;
    for (int l = 0; l < LSEQ; ++l) mx = fmaxf(mx, sZ[tid * LSEQ + l]);
    float sum = 0.0f;
    for (int l = 0; l < LSEQ; ++l) sum += __expf(sZ[tid * LSEQ + l] - mx);
    sRed[tid * 2]     = mx;
    sRed[tid * 2 + 1] = sum;
  }
  __syncthreads();
  {
    const int d = tid, h = d >> 6;            // one output feature per thread
    const float mx  = sRed[h * 2];
    const float inv = 1.0f / sRed[h * 2 + 1];
    float acc = 0.0f;
    for (int l = 0; l < LSEQ; ++l)
      acc += __expf(sZ[h * LSEQ + l] - mx) * (float)sF[l * DMODEL + d];
    out[(size_t)n * DMODEL + d] = acc * inv;
  }
}

// ---------------- host-side launch ----------------
extern "C" void kernel_launch(void* const* d_in, const int* in_sizes, int n_in,
                              void* d_out, int out_size, void* d_ws, size_t ws_size,
                              hipStream_t stream) {
  const float* tokens  = (const float*)d_in[0];
  const float* ts      = (const float*)d_in[1];
  const float* mask    = (const float*)d_in[2];
  const float* Wq      = (const float*)d_in[3];
  const float* bq      = (const float*)d_in[4];
  const float* Wk      = (const float*)d_in[5];
  const float* bk      = (const float*)d_in[6];
  const float* Wv      = (const float*)d_in[7];
  const float* bv      = (const float*)d_in[8];
  const float* Wo      = (const float*)d_in[9];
  const float* bo      = (const float*)d_in[10];
  const float* readout = (const float*)d_in[11];
  float* out = (float*)d_out;

  _Float16* w16 = (_Float16*)d_ws;            // 4 x 256 x 256 f16 = 512 KB

  convert_weights_kernel<<<(4 * DMODEL * DMODEL) / 256, 256, 0, stream>>>(
      Wq, Wk, Wv, Wo, w16);

  const size_t shmem = 157216;                // ~154 KB, within 320 KB WGP LDS
  patch_attn_pool_kernel<<<N_PATCH, 256, shmem, stream>>>(
      tokens, ts, mask, bq, bk, bv, bo, readout, w16, out);
}
